// HybridTransformerNet_65481071396232
// MI455X (gfx1250) — matrix operand order repaired and checked by reading, not compiled, not run on hardware
//
#include <hip/hip_runtime.h>

// ---------------------------------------------------------------------------
// HybridTransformerNet forward for MI455X (gfx1250, wave32, WMMA).
// - All GEMMs + both attention GEMMs on v_wmma_f32_16x16x32_bf16.
// - Weights pre-converted once per launch to bf16 [N][K]; GEMM B-tiles are
//   then streamed into LDS with GLOBAL_LOAD_ASYNC_TO_LDS_B128 (ASYNCcnt),
//   double-buffered so the copy overlaps the WMMA math.
// - Attention is fused flash-style (online softmax): the 402MB/layer score
//   matrix is never materialized.
// - All GEMM launches have M % 128 == 0 and N % 64 == 0, so the inner loop
//   and epilogue carry no bounds guards (no exec-mask churn).
// ---------------------------------------------------------------------------

typedef __bf16 v16bf __attribute__((ext_vector_type(16)));
typedef float  v8f   __attribute__((ext_vector_type(8)));
typedef float  v4f   __attribute__((ext_vector_type(4)));
typedef unsigned int v4u __attribute__((ext_vector_type(4)));

union Frag16 {               // one 16x32 (A) or 32x16 (B) bf16 WMMA fragment
    v16bf v;
    unsigned short u[16];
    v4u q[2];
};

__device__ __forceinline__ unsigned short f32_to_bf16(float f) {
    union { float f; unsigned int u; } x; x.f = f;
    unsigned int u = x.u;
    return (unsigned short)((u + 0x7FFFu + ((u >> 16) & 1u)) >> 16);  // RNE
}

__device__ __forceinline__ void cvt8(unsigned short* d, v4f a, v4f b) {
    d[0] = f32_to_bf16(a[0]); d[1] = f32_to_bf16(a[1]);
    d[2] = f32_to_bf16(a[2]); d[3] = f32_to_bf16(a[3]);
    d[4] = f32_to_bf16(b[0]); d[5] = f32_to_bf16(b[1]);
    d[6] = f32_to_bf16(b[2]); d[7] = f32_to_bf16(b[3]);
}

__device__ __forceinline__ v8f vzero8() {
    v8f z = {0.f,0.f,0.f,0.f,0.f,0.f,0.f,0.f};
    return z;
}

// LDS byte offset from a generic pointer to __shared__ memory: per the CDNA5
// aperture rule, addr[31:0] of a shared-aperture address IS the LDS offset.
__device__ __forceinline__ unsigned lds_off(const void* p) {
    return (unsigned)(unsigned long long)p;
}

// Async copy of 16 bytes global -> LDS (ASYNCcnt), SADDR + 32-bit voffset form.
__device__ __forceinline__ void async_b128(unsigned dst_lds, unsigned voff_bytes,
                                           const unsigned short* sbase) {
    asm volatile("global_load_async_to_lds_b128 %0, %1, %2 offset:0"
                 :
                 : "v"(dst_lds), "v"(voff_bytes), "s"(sbase)
                 : "memory");
}

__device__ __forceinline__ void wait_async_le1() {
    asm volatile("s_wait_asynccnt 0x1" ::: "memory");
}
__device__ __forceinline__ void wait_async_le0() {
    asm volatile("s_wait_asynccnt 0x0" ::: "memory");
}

// Load an A-fragment (16x32, row-major f32 source) straight from global.
// Lane (hl, r): row = r; K chunks {hl*8..+7} and {16+hl*8..+7} are contiguous.
// Caller guarantees the row is in range (M % 128 == 0 for all launches).
__device__ __forceinline__ void load_a_frag(Frag16& f, const float* __restrict__ A,
                                            long row, int lda, int k0, int hl) {
    const float* p = A + row * (long)lda + k0 + hl * 8;
    v4f c0 = *(const v4f*)(p);
    v4f c1 = *(const v4f*)(p + 4);
    v4f c2 = *(const v4f*)(p + 16);
    v4f c3 = *(const v4f*)(p + 20);
    cvt8(f.u + 0, c0, c1);
    cvt8(f.u + 8, c2, c3);
}

// ---------------------------------------------------------------------------
// Weight pre-pass: W[K][N] f32 -> Wt[N][K] bf16 (transposed, async-copy ready)
// ---------------------------------------------------------------------------
__global__ __launch_bounds__(256) void wcvt_kernel(const float* __restrict__ W,
                                                   unsigned short* __restrict__ Wt,
                                                   int K, int N)
{
    long idx = (long)blockIdx.x * 256 + threadIdx.x;
    if (idx < (long)K * N) {
        int n = (int)(idx % N);
        int k = (int)(idx / N);
        Wt[(long)n * K + k] = f32_to_bf16(W[idx]);
    }
}

// ---------------------------------------------------------------------------
// GEMM:  C[M,N] = act(A[M,K] @ B[K,N] + bias),  B given as bf16 Bt[N][K].
// Tile 128(M) x 64(N), 8 waves, each wave 32x32 output.
// A frags load direct from global (f32 -> bf16 in VALU); B tiles stream
// global -> LDS via async b128 copies, double buffered (copy overlaps WMMA).
// Requires M % 128 == 0, N % 64 == 0, K % 32 == 0.
// flags: bit0 = add bias, bit1 = relu
// ---------------------------------------------------------------------------
__global__ __launch_bounds__(256) void gemm_kernel(
    const float* __restrict__ A, int lda,
    const unsigned short* __restrict__ Bt,   // [N][K] bf16
    float* __restrict__ Cp, int ldc,
    const float* __restrict__ bias,
    int M, int N, int K, int flags)
{
    __shared__ alignas(16) unsigned short sBt[2][64][40];   // [buf][n][k(32)+pad]

    const int tid = threadIdx.x;
    const int w = tid >> 5, lane = tid & 31;
    const int hl = lane >> 4, r = lane & 15;
    const int waveM = w & 3, waveN = w >> 2;
    const int m0 = blockIdx.y * 128, n0 = blockIdx.x * 64;

    // async-copy assignment: 256 threads x 16B = one 64x32 bf16 tile
    const int nn = tid >> 2;                 // 0..63 : row (n) of tile
    const int cc = tid & 3;                  // 16B chunk within 64B row
    const unsigned ldsA = lds_off(&sBt[0][nn][cc * 8]);
    const unsigned ldsB = lds_off(&sBt[1][nn][cc * 8]);
    const unsigned voff = (unsigned)(((long)nn * K + cc * 8) * 2);
    const unsigned short* gbase = Bt + (long)n0 * K;

    const long arow0 = (long)(m0 + waveM * 32 + r);

    v8f acc[2][2];
    acc[0][0] = vzero8(); acc[0][1] = vzero8();
    acc[1][0] = vzero8(); acc[1][1] = vzero8();

    const int ksteps = K >> 5;               // K is a multiple of 32
    async_b128(ldsA, voff, gbase);           // prime buffer 0 (tile 0)

    for (int t = 0; t < ksteps; ++t) {
        const int k0 = t << 5;
        const int cur = t & 1;

        __syncthreads();                     // prior reads of buf cur^1 done
        if (t + 1 < ksteps) {
            async_b128(cur ? ldsA : ldsB, voff, gbase + ((t + 1) << 5));
            wait_async_le1();                // own tile t complete
        } else {
            wait_async_le0();
        }
        __syncthreads();                     // tile t visible to all waves

        Frag16 a[2];
        load_a_frag(a[0], A, arow0,      lda, k0, hl);
        load_a_frag(a[1], A, arow0 + 16, lda, k0, hl);
        // speculative prefetch of the next A chunk (dropped if out of range)
        __builtin_prefetch(A + arow0 * lda + k0 + 32, 0, 2);

        Frag16 b[2];
        #pragma unroll
        for (int ni = 0; ni < 2; ++ni) {
            int n = waveN * 32 + ni * 16 + r;
            b[ni].q[0] = *(const v4u*)&sBt[cur][n][hl * 8];
            b[ni].q[1] = *(const v4u*)&sBt[cur][n][16 + hl * 8];
        }

        #pragma unroll
        for (int mi = 0; mi < 2; ++mi)
            #pragma unroll
            for (int ni = 0; ni < 2; ++ni)
                acc[mi][ni] = __builtin_amdgcn_wmma_f32_16x16x32_bf16(
                    false, a[mi].v, false, b[ni].v, (short)0, acc[mi][ni], false, false);
    }

    // epilogue (no guards: M % 128 == 0, N % 64 == 0 for all launches)
    #pragma unroll
    for (int mi = 0; mi < 2; ++mi)
        #pragma unroll
        for (int ni = 0; ni < 2; ++ni)
            #pragma unroll
            for (int j = 0; j < 8; ++j) {
                int row = m0 + waveM * 32 + mi * 16 + j + hl * 8;
                int col = n0 + waveN * 32 + ni * 16 + r;
                float v = acc[mi][ni][j];
                if (flags & 1) v += bias[col];
                if (flags & 2) v = fmaxf(v, 0.f);
                Cp[(long)row * ldc + col] = v;
            }
}

// ---------------------------------------------------------------------------
// Fused flash attention. qkv layout: [B*S, 3*768] (q | k | v), head dk=64.
// Block = (qtile 64, head, batch); 4 waves, each wave owns 16 query rows.
// ---------------------------------------------------------------------------
__global__ __launch_bounds__(128) void attn_kernel(const float* __restrict__ qkv,
                                                   float* __restrict__ outp)
{
    const int bz = blockIdx.z, hz = blockIdx.y;
    const int q0 = blockIdx.x * 64;
    const int tid = threadIdx.x;
    const int w = tid >> 5, lane = tid & 31;
    const int hl = lane >> 4, r = lane & 15;

    __shared__ alignas(16) unsigned short sVt[64][40];      // [dk][key(32)]
    __shared__ alignas(16) unsigned short sP[4][16][40];    // per-wave P tile

    // Q fragments (kstep 0: dk 0..31, kstep 1: dk 32..63) held all kernel
    Frag16 aq[2];
    {
        const long qrow = (long)bz * 1024 + q0 + w * 16 + r;
        const float* qp = qkv + qrow * 2304 + hz * 64;
        #pragma unroll
        for (int ks = 0; ks < 2; ++ks) {
            const float* p = qp + ks * 32 + hl * 8;
            v4f c0 = *(const v4f*)(p);
            v4f c1 = *(const v4f*)(p + 4);
            v4f c2 = *(const v4f*)(p + 16);
            v4f c3 = *(const v4f*)(p + 20);
            cvt8(aq[ks].u + 0, c0, c1);
            cvt8(aq[ks].u + 8, c2, c3);
        }
    }

    float mstate[8], lstate[8];
    v8f o[4];
    #pragma unroll
    for (int j = 0; j < 8; ++j) { mstate[j] = -1e30f; lstate[j] = 0.f; }
    #pragma unroll
    for (int nf = 0; nf < 4; ++nf) o[nf] = vzero8();

    for (int kt0 = 0; kt0 < 1024; kt0 += 32) {
        __syncthreads();                      // WAR on sVt across waves
        // stage V tile transposed (32 keys x 64 dk), f32 -> bf16
        #pragma unroll
        for (int i = 0; i < 16; ++i) {
            int e   = tid + i * 128;
            int key = e >> 6, dk = e & 63;
            float v = qkv[(long)(bz * 1024 + kt0 + key) * 2304 + 1536 + hz * 64 + dk];
            sVt[dk][key] = f32_to_bf16(v);
        }
        __syncthreads();

        // ---- scores: Q(16x64) @ K^T(64x32) ----
        v8f s0 = vzero8(), s1 = vzero8();
        #pragma unroll
        for (int ks = 0; ks < 2; ++ks) {
            Frag16 bk0, bk1;
            const float* kb0 = qkv + (long)(bz * 1024 + kt0 + r) * 2304
                               + 768 + hz * 64 + ks * 32 + hl * 8;
            const float* kb1 = kb0 + (long)16 * 2304;
            {
                v4f c0 = *(const v4f*)(kb0);     v4f c1 = *(const v4f*)(kb0 + 4);
                v4f c2 = *(const v4f*)(kb0 + 16); v4f c3 = *(const v4f*)(kb0 + 20);
                cvt8(bk0.u + 0, c0, c1); cvt8(bk0.u + 8, c2, c3);
            }
            {
                v4f c0 = *(const v4f*)(kb1);     v4f c1 = *(const v4f*)(kb1 + 4);
                v4f c2 = *(const v4f*)(kb1 + 16); v4f c3 = *(const v4f*)(kb1 + 20);
                cvt8(bk1.u + 0, c0, c1); cvt8(bk1.u + 8, c2, c3);
            }
            s0 = __builtin_amdgcn_wmma_f32_16x16x32_bf16(false, aq[ks].v, false, bk0.v,
                                                         (short)0, s0, false, false);
            s1 = __builtin_amdgcn_wmma_f32_16x16x32_bf16(false, aq[ks].v, false, bk1.v,
                                                         (short)0, s1, false, false);
        }
        s0 *= 0.125f;   // 1/sqrt(64)
        s1 *= 0.125f;

        // ---- online softmax (per row j + hl*8; 16-lane reductions) ----
        #pragma unroll
        for (int j = 0; j < 8; ++j) {
            float rm = fmaxf(s0[j], s1[j]);
            #pragma unroll
            for (int m = 1; m < 16; m <<= 1) rm = fmaxf(rm, __shfl_xor(rm, m, 32));
            float nm = fmaxf(mstate[j], rm);
            float corr = __expf(mstate[j] - nm);
            mstate[j] = nm;
            float e0 = __expf(s0[j] - nm);
            float e1 = __expf(s1[j] - nm);
            s0[j] = e0; s1[j] = e1;
            float rs = e0 + e1;
            #pragma unroll
            for (int m = 1; m < 16; m <<= 1) rs += __shfl_xor(rs, m, 32);
            lstate[j] = lstate[j] * corr + rs;
            o[0][j] *= corr; o[1][j] *= corr; o[2][j] *= corr; o[3][j] *= corr;
        }

        // ---- P@V: transpose P (C-layout -> A-layout) via per-wave LDS ----
        #pragma unroll
        for (int j = 0; j < 8; ++j) {
            int row = j + hl * 8;
            sP[w][row][r]      = f32_to_bf16(s0[j]);
            sP[w][row][16 + r] = f32_to_bf16(s1[j]);
        }
        Frag16 ap;                                  // same-wave DS is in-order
        ap.q[0] = *(const v4u*)&sP[w][r][hl * 8];
        ap.q[1] = *(const v4u*)&sP[w][r][16 + hl * 8];

        #pragma unroll
        for (int nf = 0; nf < 4; ++nf) {
            Frag16 bv;
            bv.q[0] = *(const v4u*)&sVt[nf * 16 + r][hl * 8];
            bv.q[1] = *(const v4u*)&sVt[nf * 16 + r][16 + hl * 8];
            o[nf] = __builtin_amdgcn_wmma_f32_16x16x32_bf16(false, ap.v, false, bv.v,
                                                            (short)0, o[nf], false, false);
        }
    }

    // epilogue: normalize and scatter to [B*S, 768] (heads concatenated)
    #pragma unroll
    for (int nf = 0; nf < 4; ++nf)
        #pragma unroll
        for (int j = 0; j < 8; ++j) {
            long qrow = (long)bz * 1024 + q0 + w * 16 + j + hl * 8;
            outp[qrow * 768 + hz * 64 + nf * 16 + r] = o[nf][j] / lstate[j];
        }
}

// ---------------------------------------------------------------------------
// Elementwise / reduction kernels (f32)
// ---------------------------------------------------------------------------
__global__ __launch_bounds__(256) void embed_kernel(const int* __restrict__ x,
                                                    const float* __restrict__ emb,
                                                    float* __restrict__ h)
{
    int idx = blockIdx.x * 256 + threadIdx.x;       // < 8192*768
    int n = idx / 768, d = idx % 768;
    int s = n & 1023;
    int tok = x[n];
    int i2 = d & ~1;
    float dv = __expf((float)i2 * (-9.210340371976184f / 768.0f));
    float ang = (float)s * dv;
    float pe = (d & 1) ? __cosf(ang) : __sinf(ang);
    h[idx] = emb[(long)tok * 768 + d] + pe;
}

__global__ __launch_bounds__(256) void add_ln_kernel(float* __restrict__ x,
                                                     const float* __restrict__ rsd,
                                                     const float* __restrict__ g,
                                                     const float* __restrict__ b)
{
    int row = blockIdx.x, tid = threadIdx.x;
    __shared__ float sred[256];
    float v[3];
    float s1 = 0.f, s2 = 0.f;
    #pragma unroll
    for (int i = 0; i < 3; ++i) {
        int d = tid + i * 256;
        float t = x[(long)row * 768 + d] + rsd[(long)row * 768 + d];
        v[i] = t; s1 += t; s2 += t * t;
    }
    sred[tid] = s1; __syncthreads();
    for (int o = 128; o > 0; o >>= 1) { if (tid < o) sred[tid] += sred[tid + o]; __syncthreads(); }
    float mean = sred[0] * (1.f / 768.f); __syncthreads();
    sred[tid] = s2; __syncthreads();
    for (int o = 128; o > 0; o >>= 1) { if (tid < o) sred[tid] += sred[tid + o]; __syncthreads(); }
    float var = sred[0] * (1.f / 768.f) - mean * mean;
    float inv = rsqrtf(var + 1e-5f);
    #pragma unroll
    for (int i = 0; i < 3; ++i) {
        int d = tid + i * 256;
        x[(long)row * 768 + d] = (v[i] - mean) * inv * g[d] + b[d];
    }
}

__global__ __launch_bounds__(256) void pool_kernel(const float* __restrict__ h,
                                                   float* __restrict__ pooled)
{
    int idx = blockIdx.x * 256 + threadIdx.x;       // < 8*768
    int b = idx / 768, d = idx % 768;
    float s = 0.f;
    for (int t = 0; t < 1024; ++t) s += h[((long)b * 1024 + t) * 768 + d];
    pooled[idx] = s * (1.f / 1024.f);
}

__global__ __launch_bounds__(128) void cls_kernel(const float* __restrict__ pooled,
                                                  const float* __restrict__ Wc,
                                                  const float* __restrict__ bc,
                                                  float* __restrict__ outp)
{
    int t = threadIdx.x;
    if (t < 80) {
        int b = t / 10, c = t % 10;
        float s = bc[c];
        for (int d = 0; d < 768; ++d) s += pooled[b * 768 + d] * Wc[d * 10 + c];
        outp[t] = s;
    }
}

// ---------------------------------------------------------------------------
// Host launcher
// ---------------------------------------------------------------------------
extern "C" void kernel_launch(void* const* d_in, const int* in_sizes, int n_in,
                              void* d_out, int out_size, void* d_ws, size_t ws_size,
                              hipStream_t stream)
{
    constexpr int Dd = 768, Hh = 12, Ll = 6, Ff = 3072, Bb = 8, Ss = 1024;
    constexpr int NTOK = Bb * Ss;                    // 8192

    const int*   x    = (const int*)  d_in[0];
    const float* emb  = (const float*)d_in[1];
    const float* Wq   = (const float*)d_in[2];
    const float* Wk   = (const float*)d_in[3];
    const float* Wv   = (const float*)d_in[4];
    const float* Wo   = (const float*)d_in[5];
    const float* W1   = (const float*)d_in[6];
    const float* bf1  = (const float*)d_in[7];
    const float* W2   = (const float*)d_in[8];
    const float* bf2  = (const float*)d_in[9];
    const float* ln1g = (const float*)d_in[10];
    const float* ln1b = (const float*)d_in[11];
    const float* ln2g = (const float*)d_in[12];
    const float* ln2b = (const float*)d_in[13];
    const float* Wc   = (const float*)d_in[14];
    const float* bc   = (const float*)d_in[15];
    float* outp = (float*)d_out;

    // workspace: f32 activations (~176MB) + bf16 transposed weights (~85MB)
    float* h      = (float*)d_ws;
    float* tmp    = h    + (size_t)NTOK * Dd;
    float* atn    = tmp  + (size_t)NTOK * Dd;
    float* big    = atn  + (size_t)NTOK * Dd;        // qkv [NTOK,2304] / ffn1 [NTOK,3072]
    float* pooled = big  + (size_t)NTOK * Ff;
    unsigned short* wqT = (unsigned short*)(pooled + (size_t)Bb * Dd);
    unsigned short* wkT = wqT + (size_t)Ll * Dd * Dd;
    unsigned short* wvT = wkT + (size_t)Ll * Dd * Dd;
    unsigned short* woT = wvT + (size_t)Ll * Dd * Dd;
    unsigned short* w1T = woT + (size_t)Ll * Dd * Dd;   // [F][D] per layer
    unsigned short* w2T = w1T + (size_t)Ll * Ff * Dd;   // [D][F] per layer

    // ---- weight pre-pass: f32 [K][N] -> bf16 [N][K] ----
    const int gDD = (Dd * Dd) / 256;                 // 2304 blocks
    const int gDF = (Dd * Ff) / 256;                 // 9216 blocks
    for (int l = 0; l < Ll; ++l) {
        wcvt_kernel<<<gDD, 256, 0, stream>>>(Wq + (size_t)l * Dd * Dd, wqT + (size_t)l * Dd * Dd, Dd, Dd);
        wcvt_kernel<<<gDD, 256, 0, stream>>>(Wk + (size_t)l * Dd * Dd, wkT + (size_t)l * Dd * Dd, Dd, Dd);
        wcvt_kernel<<<gDD, 256, 0, stream>>>(Wv + (size_t)l * Dd * Dd, wvT + (size_t)l * Dd * Dd, Dd, Dd);
        wcvt_kernel<<<gDD, 256, 0, stream>>>(Wo + (size_t)l * Dd * Dd, woT + (size_t)l * Dd * Dd, Dd, Dd);
        wcvt_kernel<<<gDF, 256, 0, stream>>>(W1 + (size_t)l * Dd * Ff, w1T + (size_t)l * Ff * Dd, Dd, Ff);
        wcvt_kernel<<<gDF, 256, 0, stream>>>(W2 + (size_t)l * Ff * Dd, w2T + (size_t)l * Dd * Ff, Ff, Dd);
    }

    embed_kernel<<<NTOK * Dd / 256, 256, 0, stream>>>(x, emb, h);

    const dim3 g768(Dd / 64, NTOK / 128);            // (12, 64)
    const dim3 gffn(Ff / 64, NTOK / 128);            // (48, 64)
    const dim3 gattn(Ss / 64, Hh, Bb);               // (16, 12, 8)

    for (int l = 0; l < Ll; ++l) {
        // QKV into packed [NTOK, 2304] buffer
        gemm_kernel<<<g768, 256, 0, stream>>>(h, Dd, wqT + (size_t)l * Dd * Dd, big + 0 * Dd, 3 * Dd, nullptr, NTOK, Dd, Dd, 0);
        gemm_kernel<<<g768, 256, 0, stream>>>(h, Dd, wkT + (size_t)l * Dd * Dd, big + 1 * Dd, 3 * Dd, nullptr, NTOK, Dd, Dd, 0);
        gemm_kernel<<<g768, 256, 0, stream>>>(h, Dd, wvT + (size_t)l * Dd * Dd, big + 2 * Dd, 3 * Dd, nullptr, NTOK, Dd, Dd, 0);

        attn_kernel<<<gattn, 128, 0, stream>>>(big, atn);

        gemm_kernel<<<g768, 256, 0, stream>>>(atn, Dd, woT + (size_t)l * Dd * Dd, tmp, Dd, nullptr, NTOK, Dd, Dd, 0);
        add_ln_kernel<<<NTOK, 256, 0, stream>>>(h, tmp, ln1g + l * Dd, ln1b + l * Dd);

        gemm_kernel<<<gffn, 256, 0, stream>>>(h, Dd, w1T + (size_t)l * Ff * Dd, big, Ff, bf1 + l * Ff, NTOK, Ff, Dd, 3);
        gemm_kernel<<<g768, 256, 0, stream>>>(big, Ff, w2T + (size_t)l * Dd * Ff, tmp, Dd, bf2 + l * Dd, NTOK, Dd, Ff, 1);
        add_ln_kernel<<<NTOK, 256, 0, stream>>>(h, tmp, ln2g + l * Dd, ln2b + l * Dd);
    }

    pool_kernel<<<(Bb * Dd) / 256, 256, 0, stream>>>(h, pooled);
    cls_kernel<<<1, 128, 0, stream>>>(pooled, Wc, bc, outp);
}
